// TransformerBlock_54606214201758
// MI455X (gfx1250) — matrix operand order repaired and checked by reading, not compiled, not run on hardware
//
#include <hip/hip_runtime.h>
#include <math.h>
#include <stdint.h>

typedef __bf16 bf16_t;
typedef __bf16 v16bf __attribute__((ext_vector_type(16)));
typedef float  v8f   __attribute__((ext_vector_type(8)));

#define WMMA_BF16(a, b, c) \
  __builtin_amdgcn_wmma_f32_16x16x32_bf16(false, (a), false, (b), (short)0, (c), false, false)

// ---- model dims (fixed) ----
#define BB   2
#define TT   2048
#define DD   2048
#define HH   16
#define KHH  4
#define FF   8192
#define HDD  128
#define MR   (BB * TT)   // 4096 rows

// =====================================================================
// gfx1250 async global->LDS path (guarded; falls back to sync copies)
// Builtin signature: (int4 AS1* src, int4 AS3* dst, imm offset, imm cpol)
// =====================================================================
#if defined(__AMDGCN__) && \
    __has_builtin(__builtin_amdgcn_global_load_async_to_lds_b128) && \
    __has_builtin(__builtin_amdgcn_s_wait_asynccnt)
#define HAS_ASYNC 1
#else
#define HAS_ASYNC 0
#endif

__device__ __forceinline__ void cp16_to_lds(bf16_t* lds, const bf16_t* g) {
#if HAS_ASYNC
  typedef int i32x4 __attribute__((vector_size(16)));
  typedef __attribute__((address_space(1))) i32x4* gp_t;
  typedef __attribute__((address_space(3))) i32x4* lp_t;
  __builtin_amdgcn_global_load_async_to_lds_b128(
      (gp_t)(uintptr_t)g, (lp_t)(uint32_t)(uintptr_t)lds, 0, 0);
#else
  *(uint4*)lds = *(const uint4*)g;
#endif
}

__device__ __forceinline__ void async_wait0() {
#if HAS_ASYNC
  __builtin_amdgcn_s_wait_asynccnt(0);
#endif
}

// =====================================================================
// WMMA fragment loaders (CDNA5 16-bit layouts, wave32)
// A (16xK chunk of 32): lane m = M, element e -> K = 16*(e/8) + 8*hi + e%8
// B (Kx16 chunk of 32): lane n = N, element e -> K = 16*(e/8) + 8*hi + e%8
// C/D f32: lane n = N, vgpr r -> M = r + 8*hi
// =====================================================================
__device__ __forceinline__ v16bf frag_a(const bf16_t* p, int ld) {
  int lane = threadIdx.x & 31;
  int m = lane & 15, hi = lane >> 4;
  const bf16_t* row = p + m * ld + hi * 8;
  v16bf a;
#pragma unroll
  for (int g = 0; g < 2; ++g)
#pragma unroll
    for (int j = 0; j < 8; ++j)
      a[g * 8 + j] = row[g * 16 + j];
  return a;
}

__device__ __forceinline__ v16bf frag_b(const bf16_t* p, int ld) {
  int lane = threadIdx.x & 31;
  int n = lane & 15, hi = lane >> 4;
  const bf16_t* col = p + n * ld + hi * 8;
  v16bf b;
#pragma unroll
  for (int e = 0; e < 16; ++e) b[e] = col[(e >> 3) * 16 + (e & 7)];
  return b;
}

// =====================================================================
// fp32 -> bf16 convert
// =====================================================================
__global__ void k_cvt(const float* __restrict__ w, bf16_t* __restrict__ o, int n) {
  int i = blockIdx.x * 256 + threadIdx.x;
  if (i < n) o[i] = (bf16_t)w[i];
}

// =====================================================================
// LayerNorm over DD, one block per row. OUTBF: 1 -> bf16 out, 0 -> f32
// =====================================================================
template <int OUTBF>
__global__ void __launch_bounds__(256)
k_ln(const float* __restrict__ x, const float* __restrict__ g,
     const float* __restrict__ bta, void* __restrict__ out) {
  __shared__ float red[256];
  int row = blockIdx.x, tid = threadIdx.x;
  const float* xr = x + (size_t)row * DD;

  float s = 0.f;
  for (int i = tid; i < DD; i += 256) s += xr[i];
  red[tid] = s;
  __syncthreads();
  for (int st = 128; st > 0; st >>= 1) {
    if (tid < st) red[tid] += red[tid + st];
    __syncthreads();
  }
  float mean = red[0] * (1.0f / DD);
  __syncthreads();

  float s2 = 0.f;
  for (int i = tid; i < DD; i += 256) {
    float d = xr[i] - mean;
    s2 += d * d;
  }
  red[tid] = s2;
  __syncthreads();
  for (int st = 128; st > 0; st >>= 1) {
    if (tid < st) red[tid] += red[tid + st];
    __syncthreads();
  }
  float inv = rsqrtf(red[0] * (1.0f / DD) + 1e-5f);

  for (int i = tid; i < DD; i += 256) {
    float y = (xr[i] - mean) * inv * g[i] + bta[i];
    if (OUTBF)
      ((bf16_t*)out)[(size_t)row * DD + i] = (bf16_t)y;
    else
      ((float*)out)[(size_t)row * DD + i] = y;
  }
}

// =====================================================================
// Tiled WMMA GEMM, double-buffered LDS + async A staging.
// C[M,N] = A[M,K](bf16) @ W[K,N](bf16) + bias (+res)
// BM=128, BN=64, BK=32; 8 waves, each computes 32x32 (4 wmma / k-step)
// =====================================================================
#define BM  128
#define BN  64
#define BK  32
#define LDA 40
#define LDB 40

template <int OUTBF>
__global__ void __launch_bounds__(256)
k_gemm(const bf16_t* __restrict__ A, const bf16_t* __restrict__ W,
       const float* __restrict__ bias, const float* __restrict__ res,
       void* __restrict__ out, int M, int N, int K) {
  __shared__ __align__(16) bf16_t As[2][BM * LDA];
  __shared__ __align__(16) bf16_t Bs[2][BN * LDB];

  int tid = threadIdx.x;
  int m0 = blockIdx.y * BM;
  int n0 = blockIdx.x * BN;
  int w = tid >> 5;
  int wm = w & 3, wn = w >> 2;

  v8f c[2][2] = {};

  auto stage = [&](int buf, int k0) {
#pragma unroll
    for (int it = 0; it < 2; ++it) {
      int idx = it * 256 + tid;
      int row = idx >> 2, cc = idx & 3;
      cp16_to_lds(&As[buf][row * LDA + cc * 8],
                  A + (size_t)(m0 + row) * K + k0 + cc * 8);
    }
    int kr = tid >> 3, nc = tid & 7;
    uint4 t = *(const uint4*)(W + (size_t)(k0 + kr) * N + n0 + nc * 8);
    const bf16_t* tv = (const bf16_t*)&t;
#pragma unroll
    for (int j = 0; j < 8; ++j) Bs[buf][(nc * 8 + j) * LDB + kr] = tv[j];
  };

  int nk = K / BK;
  stage(0, 0);
  for (int kk = 0; kk < nk; ++kk) {
    int cur = kk & 1;
    async_wait0();
    __syncthreads();
    if (kk + 1 < nk) stage(cur ^ 1, (kk + 1) * BK);

    const bf16_t* Ab = As[cur];
    const bf16_t* Bb = Bs[cur];
    v16bf a0 = frag_a(Ab + (wm * 32) * LDA, LDA);
    v16bf a1 = frag_a(Ab + (wm * 32 + 16) * LDA, LDA);
    v16bf b0 = frag_b(Bb + (wn * 32) * LDB, LDB);
    v16bf b1 = frag_b(Bb + (wn * 32 + 16) * LDB, LDB);
    c[0][0] = WMMA_BF16(a0, b0, c[0][0]);
    c[0][1] = WMMA_BF16(a0, b1, c[0][1]);
    c[1][0] = WMMA_BF16(a1, b0, c[1][0]);
    c[1][1] = WMMA_BF16(a1, b1, c[1][1]);
  }

  int lane = tid & 31, n = lane & 15, hi = lane >> 4;
#pragma unroll
  for (int mi = 0; mi < 2; ++mi)
#pragma unroll
    for (int ni = 0; ni < 2; ++ni)
#pragma unroll
      for (int r = 0; r < 8; ++r) {
        int gm = m0 + wm * 32 + mi * 16 + r + hi * 8;
        int gn = n0 + wn * 32 + ni * 16 + n;
        size_t idx = (size_t)gm * N + gn;
        float val = c[mi][ni][r] + bias[gn];
        if (OUTBF) {
          ((bf16_t*)out)[idx] = (bf16_t)val;
        } else {
          if (res) val += res[idx];
          ((float*)out)[idx] = val;
        }
      }
}

// =====================================================================
// Fused dual-GEMM SwiGLU: out = silu-ish(A@W1+b1) * (A@W2+b2)  (bf16)
// =====================================================================
__global__ void __launch_bounds__(256)
k_gemm_swiglu(const bf16_t* __restrict__ A, const bf16_t* __restrict__ W1,
              const bf16_t* __restrict__ W2, const float* __restrict__ bias1,
              const float* __restrict__ bias2, bf16_t* __restrict__ out,
              int M, int N, int K) {
  __shared__ __align__(16) bf16_t As[2][BM * LDA];
  __shared__ __align__(16) bf16_t B1s[2][BN * LDB];
  __shared__ __align__(16) bf16_t B2s[2][BN * LDB];

  int tid = threadIdx.x;
  int m0 = blockIdx.y * BM;
  int n0 = blockIdx.x * BN;
  int w = tid >> 5;
  int wm = w & 3, wn = w >> 2;

  v8f c1[2][2] = {};
  v8f c2[2][2] = {};

  auto stage = [&](int buf, int k0) {
#pragma unroll
    for (int it = 0; it < 2; ++it) {
      int idx = it * 256 + tid;
      int row = idx >> 2, cc = idx & 3;
      cp16_to_lds(&As[buf][row * LDA + cc * 8],
                  A + (size_t)(m0 + row) * K + k0 + cc * 8);
    }
    int kr = tid >> 3, nc = tid & 7;
    uint4 t1 = *(const uint4*)(W1 + (size_t)(k0 + kr) * N + n0 + nc * 8);
    uint4 t2 = *(const uint4*)(W2 + (size_t)(k0 + kr) * N + n0 + nc * 8);
    const bf16_t* tv1 = (const bf16_t*)&t1;
    const bf16_t* tv2 = (const bf16_t*)&t2;
#pragma unroll
    for (int j = 0; j < 8; ++j) {
      B1s[buf][(nc * 8 + j) * LDB + kr] = tv1[j];
      B2s[buf][(nc * 8 + j) * LDB + kr] = tv2[j];
    }
  };

  int nk = K / BK;
  stage(0, 0);
  for (int kk = 0; kk < nk; ++kk) {
    int cur = kk & 1;
    async_wait0();
    __syncthreads();
    if (kk + 1 < nk) stage(cur ^ 1, (kk + 1) * BK);

    const bf16_t* Ab = As[cur];
    v16bf a0 = frag_a(Ab + (wm * 32) * LDA, LDA);
    v16bf a1 = frag_a(Ab + (wm * 32 + 16) * LDA, LDA);
    v16bf p0 = frag_b(B1s[cur] + (wn * 32) * LDB, LDB);
    v16bf p1 = frag_b(B1s[cur] + (wn * 32 + 16) * LDB, LDB);
    v16bf q0 = frag_b(B2s[cur] + (wn * 32) * LDB, LDB);
    v16bf q1 = frag_b(B2s[cur] + (wn * 32 + 16) * LDB, LDB);
    c1[0][0] = WMMA_BF16(a0, p0, c1[0][0]);
    c1[0][1] = WMMA_BF16(a0, p1, c1[0][1]);
    c1[1][0] = WMMA_BF16(a1, p0, c1[1][0]);
    c1[1][1] = WMMA_BF16(a1, p1, c1[1][1]);
    c2[0][0] = WMMA_BF16(a0, q0, c2[0][0]);
    c2[0][1] = WMMA_BF16(a0, q1, c2[0][1]);
    c2[1][0] = WMMA_BF16(a1, q0, c2[1][0]);
    c2[1][1] = WMMA_BF16(a1, q1, c2[1][1]);
  }

  int lane = tid & 31, n = lane & 15, hi = lane >> 4;
#pragma unroll
  for (int mi = 0; mi < 2; ++mi)
#pragma unroll
    for (int ni = 0; ni < 2; ++ni)
#pragma unroll
      for (int r = 0; r < 8; ++r) {
        int gm = m0 + wm * 32 + mi * 16 + r + hi * 8;
        int gn = n0 + wn * 32 + ni * 16 + n;
        float gp = c1[mi][ni][r] + bias1[gn];
        float gc = fminf(fmaxf(gp, -10.f), 10.f);
        float gate = gp * (1.f / (1.f + __expf(-gc)));
        float hv = c2[mi][ni][r] + bias2[gn];
        out[(size_t)gm * N + gn] = (bf16_t)(gate * hv);
      }
}

// =====================================================================
// Flash attention (GQA, causal). Block = (b, head, 64-query tile).
// q: [B,T,H,HD] bf16; k,v: [B,T,KH,HD] bf16; ctx: [B,T,H,HD] bf16.
// =====================================================================
#define LQK 136   // 128 + 8 pad (keeps 16B row alignment)
#define LVT 68    // 64 + 4 pad
#define LSS 66
#define LPS 66

#define ATTN_SMEM (64 * LQK * 2 * 2 /*Qs+Ks*/ + 128 * LVT * 2 /*Vt*/ + \
                   64 * LSS * 4 /*Ss*/ + 64 * LPS * 2 /*Ps*/ + 3 * 64 * 4)

__global__ void __launch_bounds__(256)
k_attn(const bf16_t* __restrict__ q, const bf16_t* __restrict__ kk,
       const bf16_t* __restrict__ vv, bf16_t* __restrict__ ctx) {
  extern __shared__ char smem[];
  bf16_t* Qs = (bf16_t*)smem;               // 64 x LQK
  bf16_t* Ks = Qs + 64 * LQK;               // 64 x LQK
  bf16_t* Vt = Ks + 64 * LQK;               // 128 x LVT (transposed V)
  float*  Ss = (float*)(Vt + 128 * LVT);    // 64 x LSS
  bf16_t* Ps = (bf16_t*)(Ss + 64 * LSS);    // 64 x LPS
  float*  mrow = (float*)(Ps + 64 * LPS);
  float*  lrow = mrow + 64;
  float*  arow = lrow + 64;

  int tid = threadIdx.x;
  int qt = blockIdx.x, h = blockIdx.y, b = blockIdx.z;
  int kvh = h >> 2;  // G = 4 query heads per kv head
  int qbase = qt * 64;
  int w = tid >> 5, lane = tid & 31;
  int qsub = w & 3, wh = w >> 2;  // wh: key-half for S, d-half for O
  int n = lane & 15, hi = lane >> 4;
  const float scale = 0.08838834764831845f;  // 1/sqrt(128)

  // load Q tile 64x128 (bf16) via async path
#pragma unroll
  for (int it = 0; it < 4; ++it) {
    int idx = it * 256 + tid;
    int row = idx >> 4, dc = idx & 15;
    cp16_to_lds(Qs + row * LQK + dc * 8,
                q + ((size_t)(b * TT + qbase + row) * HH + h) * HDD + dc * 8);
  }
  if (tid < 64) {
    mrow[tid] = -3e38f;
    lrow[tid] = 0.f;
  }
  v8f o[4] = {};
  async_wait0();
  __syncthreads();

  for (int kt = 0; kt <= qt; ++kt) {
    int kb = kt * 64;
    // load K tile 64x128 via async path
#pragma unroll
    for (int it = 0; it < 4; ++it) {
      int idx = it * 256 + tid;
      int row = idx >> 4, dc = idx & 15;
      cp16_to_lds(Ks + row * LQK + dc * 8,
                  kk + ((size_t)(b * TT + kb + row) * KHH + kvh) * HDD + dc * 8);
    }
    // load V tile transposed: Vt[d][key] (needs transpose -> regular path)
#pragma unroll
    for (int it = 0; it < 4; ++it) {
      int idx = it * 256 + tid;
      int key = idx >> 4, dc = idx & 15;
      uint4 t = *(const uint4*)(vv + ((size_t)(b * TT + kb + key) * KHH + kvh) * HDD + dc * 8);
      const bf16_t* tv = (const bf16_t*)&t;
#pragma unroll
      for (int j = 0; j < 8; ++j) Vt[(dc * 8 + j) * LVT + key] = tv[j];
    }
    async_wait0();
    __syncthreads();

    // S = Q K^T : this wave computes 16q x 32k
    v8f s[2] = {};
#pragma unroll
    for (int dchunk = 0; dchunk < 4; ++dchunk) {
      v16bf a = frag_a(Qs + qsub * 16 * LQK + dchunk * 32, LQK);
      v16bf kb0 = frag_b(Ks + (wh * 32) * LQK + dchunk * 32, LQK);
      v16bf kb1 = frag_b(Ks + (wh * 32 + 16) * LQK + dchunk * 32, LQK);
      s[0] = WMMA_BF16(a, kb0, s[0]);
      s[1] = WMMA_BF16(a, kb1, s[1]);
    }
    // scale, clip(+-100), causal mask, store to Ss
#pragma unroll
    for (int ni = 0; ni < 2; ++ni)
#pragma unroll
      for (int r = 0; r < 8; ++r) {
        int qrow = qsub * 16 + r + hi * 8;
        int kcol = wh * 32 + ni * 16 + n;
        float sv = s[ni][r] * scale;
        sv = fminf(fmaxf(sv, -100.f), 100.f);
        if (kb + kcol > qbase + qrow) sv = -3e38f;
        Ss[qrow * LSS + kcol] = sv;
      }
    __syncthreads();

    // streaming softmax per row
    if (tid < 64) {
      float mold = mrow[tid];
      float mx = mold;
      for (int j = 0; j < 64; ++j) mx = fmaxf(mx, Ss[tid * LSS + j]);
      float alpha = __expf(mold - mx);
      float sum = 0.f;
      for (int j = 0; j < 64; ++j) {
        float p = __expf(Ss[tid * LSS + j] - mx);
        Ps[tid * LPS + j] = (bf16_t)p;
        sum += p;
      }
      mrow[tid] = mx;
      lrow[tid] = lrow[tid] * alpha + sum;
      arow[tid] = alpha;
    }
    __syncthreads();

    // rescale O, accumulate P @ V (this wave: 16q x 64d)
#pragma unroll
    for (int dj = 0; dj < 4; ++dj)
#pragma unroll
      for (int r = 0; r < 8; ++r)
        o[dj][r] *= arow[qsub * 16 + r + hi * 8];
#pragma unroll
    for (int kc = 0; kc < 2; ++kc) {
      v16bf a = frag_a(Ps + qsub * 16 * LPS + kc * 32, LPS);
#pragma unroll
      for (int dj = 0; dj < 4; ++dj) {
        v16bf vb = frag_b(Vt + (wh * 64 + dj * 16) * LVT + kc * 32, LVT);
        o[dj] = WMMA_BF16(a, vb, o[dj]);
      }
    }
    __syncthreads();
  }

  // normalize and write ctx (bf16)
#pragma unroll
  for (int dj = 0; dj < 4; ++dj)
#pragma unroll
    for (int r = 0; r < 8; ++r) {
      int qrow = qsub * 16 + r + hi * 8;
      float val = o[dj][r] / lrow[qrow];
      int col = wh * 64 + dj * 16 + n;
      ctx[((size_t)(b * TT + qbase + qrow) * HH + h) * HDD + col] = (bf16_t)val;
    }
}

// =====================================================================
// host side
// =====================================================================
extern "C" void kernel_launch(void* const* d_in, const int* in_sizes, int n_in,
                              void* d_out, int out_size, void* d_ws, size_t ws_size,
                              hipStream_t stream) {
  (void)in_sizes; (void)n_in; (void)out_size; (void)ws_size;
  const float* x    = (const float*)d_in[0];
  const float* Wq   = (const float*)d_in[2];
  const float* bq   = (const float*)d_in[3];
  const float* Wk   = (const float*)d_in[4];
  const float* bk   = (const float*)d_in[5];
  const float* Wv   = (const float*)d_in[6];
  const float* bv   = (const float*)d_in[7];
  const float* Wo   = (const float*)d_in[8];
  const float* bo   = (const float*)d_in[9];
  const float* ln1g = (const float*)d_in[10];
  const float* ln1b = (const float*)d_in[11];
  const float* ln2g = (const float*)d_in[12];
  const float* ln2b = (const float*)d_in[13];
  const float* flng = (const float*)d_in[14];
  const float* flnb = (const float*)d_in[15];
  const float* W1   = (const float*)d_in[16];
  const float* b1   = (const float*)d_in[17];
  const float* W2   = (const float*)d_in[18];
  const float* b2   = (const float*)d_in[19];
  const float* W3   = (const float*)d_in[20];
  const float* b3   = (const float*)d_in[21];

  char* ws = (char*)d_ws;
  size_t off = 0;
  auto alloc = [&](size_t bytes) -> void* {
    void* p = ws + off;
    off = (off + bytes + 255) & ~(size_t)255;
    return p;
  };

  bf16_t* Wq_b  = (bf16_t*)alloc((size_t)DD * DD * 2);
  bf16_t* Wk_b  = (bf16_t*)alloc((size_t)DD * 512 * 2);
  bf16_t* Wv_b  = (bf16_t*)alloc((size_t)DD * 512 * 2);
  bf16_t* Wo_b  = (bf16_t*)alloc((size_t)DD * DD * 2);
  bf16_t* W1_b  = (bf16_t*)alloc((size_t)DD * FF * 2);
  bf16_t* W2_b  = (bf16_t*)alloc((size_t)DD * FF * 2);
  bf16_t* W3_b  = (bf16_t*)alloc((size_t)FF * DD * 2);
  bf16_t* h_b   = (bf16_t*)alloc((size_t)MR * DD * 2);
  bf16_t* q_b   = (bf16_t*)alloc((size_t)MR * DD * 2);
  bf16_t* k_b   = (bf16_t*)alloc((size_t)MR * 512 * 2);
  bf16_t* v_b   = (bf16_t*)alloc((size_t)MR * 512 * 2);
  bf16_t* ctx_b = (bf16_t*)alloc((size_t)MR * DD * 2);
  float*  x1    = (float*)alloc((size_t)MR * DD * 4);
  float*  x2    = (float*)alloc((size_t)MR * DD * 4);
  bf16_t* h2_b  = (bf16_t*)alloc((size_t)MR * DD * 2);
  bf16_t* gh_b  = (bf16_t*)alloc((size_t)MR * FF * 2);

  auto cvt = [&](const float* src, bf16_t* dst, int nelem) {
    k_cvt<<<dim3((nelem + 255) / 256), 256, 0, stream>>>(src, dst, nelem);
  };
  cvt(Wq, Wq_b, DD * DD);
  cvt(Wk, Wk_b, DD * 512);
  cvt(Wv, Wv_b, DD * 512);
  cvt(Wo, Wo_b, DD * DD);
  cvt(W1, W1_b, DD * FF);
  cvt(W2, W2_b, DD * FF);
  cvt(W3, W3_b, FF * DD);

  // h = LN1(x) -> bf16
  k_ln<1><<<MR, 256, 0, stream>>>(x, ln1g, ln1b, h_b);

  // q/k/v projections (bf16 outputs)
  k_gemm<1><<<dim3(DD / BN, MR / BM), 256, 0, stream>>>(h_b, Wq_b, bq, nullptr, q_b, MR, DD, DD);
  k_gemm<1><<<dim3(512 / BN, MR / BM), 256, 0, stream>>>(h_b, Wk_b, bk, nullptr, k_b, MR, 512, DD);
  k_gemm<1><<<dim3(512 / BN, MR / BM), 256, 0, stream>>>(h_b, Wv_b, bv, nullptr, v_b, MR, 512, DD);

  // flash attention -> ctx bf16
  k_attn<<<dim3(TT / 64, HH, BB), 256, ATTN_SMEM, stream>>>(q_b, k_b, v_b, ctx_b);

  // x1 = x + ctx @ Wo + bo (fp32)
  k_gemm<0><<<dim3(DD / BN, MR / BM), 256, 0, stream>>>(ctx_b, Wo_b, bo, x, x1, MR, DD, DD);

  // x2 = LN2(x1) fp32 ; h2 = fLN(x2) bf16
  k_ln<0><<<MR, 256, 0, stream>>>(x1, ln2g, ln2b, x2);
  k_ln<1><<<MR, 256, 0, stream>>>(x2, flng, flnb, h2_b);

  // gh = swiglu(h2 @ W1 + b1, h2 @ W2 + b2) bf16
  k_gemm_swiglu<<<dim3(FF / BN, MR / BM), 256, 0, stream>>>(h2_b, W1_b, W2_b, b1, b2, gh_b, MR, FF, DD);

  // out = x2 + gh @ W3 + b3 (fp32)
  k_gemm<0><<<dim3(DD / BN, MR / BM), 256, 0, stream>>>(gh_b, W3_b, b3, x2, d_out, MR, DD, FF);
}
